// UAMFDetFusion_75453985457478
// MI455X (gfx1250) — compile-verified, matrix operand order checked by reading, not run on hardware
//
#include <hip/hip_runtime.h>
#include <math.h>

typedef _Float16 v16h __attribute__((ext_vector_type(16)));
typedef _Float16 v8h  __attribute__((ext_vector_type(8)));
typedef float    v8f  __attribute__((ext_vector_type(8)));

#define Bc   8
#define Cc   256
#define Hc   64
#define Wc   64
#define HWc  4096
#define Kc   8
#define NHc  4
#define HIDc 16

// ---- ordered-uint encoding for float atomic max (handles negatives) ----
__device__ __forceinline__ unsigned enc_f32(float f) {
  unsigned u = __float_as_uint(f);
  return (u & 0x80000000u) ? ~u : (u | 0x80000000u);
}
__device__ __forceinline__ float dec_f32(unsigned e) {
  unsigned u = (e & 0x80000000u) ? (e & 0x7FFFFFFFu) : ~e;
  return __uint_as_float(u);
}

// ---- WMMA fragment loader: 16 halves for one lane of a 16x32 (or 32x16) tile.
// base points at (row*256 + kbase) halves; layout per CDNA5 ISA:
// lane<16: K = kbase+0..7 and kbase+16..23 ; lane>=16: kbase already shifted by 8.
__device__ __forceinline__ v16h load_frag(const _Float16* base) {
  const v8h* p = (const v8h*)base;
  v8h lo = p[0];
  v8h hi = p[2];
  v16h r;
#pragma unroll
  for (int i = 0; i < 8; ++i) { r[i] = lo[i]; r[8 + i] = hi[i]; }
  return r;
}

// D = Wg(256x256, f16 row-major) * Xl(256x16, f16 pixel-major [n][k]) -> Yl f32 [n][m]
__device__ __forceinline__ void gemm256x16(const _Float16* __restrict__ Wg,
                                           const _Float16* Xl, float* Yl,
                                           int wave, int lane) {
  const int n    = lane & 15;
  const int hi   = (lane >> 4) & 1;
  const int kofs = hi ? 8 : 0;
#pragma unroll
  for (int t = 0; t < 2; ++t) {
    const int mt = wave + t * 8;
    v8f acc = {};
#pragma unroll
    for (int kb = 0; kb < 8; ++kb) {
      const int kbase = kb * 32 + kofs;
      v16h a = load_frag(Wg + (size_t)(mt * 16 + n) * 256 + kbase);
      v16h b = load_frag(Xl + n * 256 + kbase);
      acc = __builtin_amdgcn_wmma_f32_16x16x32_f16(false, a, false, b,
                                                   (short)0, acc, false, false);
    }
#pragma unroll
    for (int r = 0; r < 8; ++r) {
      const int m = r + (hi ? 8 : 0);
      Yl[n * 256 + mt * 16 + m] = acc[r];
    }
  }
}

// ---------------- kernel 0: weight convert + reduction init ----------------
__global__ void __launch_bounds__(256)
k0_prep(const float* qw, const float* kw, const float* ow,
        _Float16* qw16, _Float16* kw16, _Float16* ow16,
        float* sum_ws, unsigned* maxe_ws) {
  int i = blockIdx.x * 256 + threadIdx.x;
  if (i < Cc * Cc) {
    qw16[i] = (_Float16)qw[i];
    kw16[i] = (_Float16)kw[i];
    ow16[i] = (_Float16)ow[i];
  }
  if (i < Bc * Cc) {
    sum_ws[i]  = 0.0f;
    maxe_ws[i] = enc_f32(-3.402823466e38f);
  }
}

// ---------------- kernel 1: offset conv3x3 + tanh + grid -> (ix,iy) --------
__global__ void __launch_bounds__(256)
k1_offsets(const float* __restrict__ xm, const float* __restrict__ cw,
           const float* __restrict__ cb, float* __restrict__ grid) {
  int p  = blockIdx.x * 256 + threadIdx.x;   // over B*HW
  int b  = p >> 12;
  int hw = p & 4095;
  int h  = hw >> 6, w = hw & 63;
  float acc[16];
#pragma unroll
  for (int i = 0; i < 16; ++i) acc[i] = 0.0f;
  const float* xb = xm + (size_t)b * Cc * HWc;
  for (int c = 0; c < Cc; ++c) {
    const float* xc = xb + c * HWc;
#pragma unroll
    for (int dh = -1; dh <= 1; ++dh) {
      int hh = h + dh;
      if ((unsigned)hh >= (unsigned)Hc) continue;
#pragma unroll
      for (int dw = -1; dw <= 1; ++dw) {
        int ww = w + dw;
        if ((unsigned)ww >= (unsigned)Wc) continue;
        float v  = xc[hh * Wc + ww];
        int tap  = (dh + 1) * 3 + (dw + 1);
#pragma unroll
        for (int oc = 0; oc < 16; ++oc)
          acc[oc] = fmaf(v, cw[(oc * Cc + c) * 9 + tap], acc[oc]);
      }
    }
  }
  const float sx = 2.0f / Wc, sy = 2.0f / Hc;
  float bx = (w + 0.5f) * sx - 1.0f;
  float by = (h + 0.5f) * sy - 1.0f;
#pragma unroll
  for (int k = 0; k < Kc; ++k) {
    float ox = tanhf(acc[2 * k]     + cb[2 * k]);
    float oy = tanhf(acc[2 * k + 1] + cb[2 * k + 1]);
    float gx = fminf(fmaxf(bx + ox * sx, -1.0f), 1.0f);
    float gy = fminf(fmaxf(by + oy * sy, -1.0f), 1.0f);
    size_t o = (((size_t)(b * Kc + k) * HWc) + hw) * 2;
    grid[o + 0] = (gx + 1.0f) * 0.5f * Wc - 0.5f;   // ix
    grid[o + 1] = (gy + 1.0f) * 0.5f * Hc - 0.5f;   // iy
  }
}

// ---------------- kernel 2: fused deformable attention (WMMA) --------------
// one block = one (b, 16-pixel tile); 8 waves
__global__ void __launch_bounds__(256)
k2_main(const float* __restrict__ xm, const float* __restrict__ xa,
        const _Float16* __restrict__ qw16, const _Float16* __restrict__ kw16,
        const _Float16* __restrict__ ow16, const float* __restrict__ grid,
        float* __restrict__ diff_ws, float* __restrict__ sum_ws,
        unsigned* __restrict__ maxe_ws) {
  extern __shared__ char smem_raw[];
  _Float16* xm16   = (_Float16*)smem_raw;                                  //  8 KB
  float*    q_lds  = (float*)   (smem_raw + 8192);                         // 16 KB
  _Float16* s_lds  = (_Float16*)(smem_raw + 8192 + 16384);                 // 64 KB
  float*    kk_lds = (float*)   (smem_raw + 8192 + 16384 + 65536);         // 16 KB
  _Float16* al16   = (_Float16*)(smem_raw + 8192 + 16384 + 65536 + 16384); //  8 KB
  float*    sim    = (float*)   (smem_raw + 8192 + 16384 + 65536 + 16384 + 8192); // 2 KB
  float*    wk     = sim + Kc * NHc * 16;                                  // 0.5 KB

  const int t    = threadIdx.x;
  const int wave = t >> 5, lane = t & 31;
  const int b    = blockIdx.x >> 8;
  const int tile = blockIdx.x & 255;
  const int pixbase = tile * 16;

  const float* xmb = xm + (size_t)b * Cc * HWc;
  const float* xab = xa + (size_t)b * Cc * HWc;

  // stage x_main tile -> f16 LDS [pix][ch]
  {
    const int pix = t & 15, cbk = (t >> 4) * 16;
#pragma unroll
    for (int c = 0; c < 16; ++c)
      xm16[pix * 256 + cbk + c] = (_Float16)xmb[(size_t)(cbk + c) * HWc + pixbase + pix];
  }
  __syncthreads();

  // q = q_w * x_main  (WMMA)
  gemm256x16(qw16, xm16, q_lds, wave, lane);
  __syncthreads();

  // per-sample: gather + k GEMM + sim
  for (int k = 0; k < Kc; ++k) {
    {
      const int pix = t & 15, cbk = (t >> 4) * 16;
      const int pg  = pixbase + pix;
      size_t go = (((size_t)(b * Kc + k) * HWc) + pg) * 2;
      float ix = grid[go + 0], iy = grid[go + 1];
      float x0f = floorf(ix), y0f = floorf(iy);
      int x0 = (int)x0f, y0 = (int)y0f;
      float wx = ix - x0f, wy = iy - y0f;
      bool vx0 = (x0 >= 0) && (x0 < Wc);
      bool vx1 = (x0 + 1 >= 0) && (x0 + 1 < Wc);
      bool vy0 = (y0 >= 0) && (y0 < Hc);
      bool vy1 = (y0 + 1 >= 0) && (y0 + 1 < Hc);
      float m00 = (vx0 && vy0) ? (1 - wx) * (1 - wy) : 0.0f;
      float m10 = (vx1 && vy0) ? wx * (1 - wy)       : 0.0f;
      float m01 = (vx0 && vy1) ? (1 - wx) * wy       : 0.0f;
      float m11 = (vx1 && vy1) ? wx * wy             : 0.0f;
      int cx0 = x0 < 0 ? 0 : (x0 > Wc - 1 ? Wc - 1 : x0);
      int cx1 = x0 + 1 < 0 ? 0 : (x0 + 1 > Wc - 1 ? Wc - 1 : x0 + 1);
      int cy0 = y0 < 0 ? 0 : (y0 > Hc - 1 ? Hc - 1 : y0);
      int cy1 = y0 + 1 < 0 ? 0 : (y0 + 1 > Hc - 1 ? Hc - 1 : y0 + 1);
      int i00 = cy0 * Wc + cx0, i10 = cy0 * Wc + cx1;
      int i01 = cy1 * Wc + cx0, i11 = cy1 * Wc + cx1;
      _Float16* sd = s_lds + k * 4096 + pix * 256 + cbk;
#pragma unroll
      for (int c = 0; c < 16; ++c) {
        const float* p = xab + (size_t)(cbk + c) * HWc;
        float v = p[i00] * m00 + p[i10] * m10 + p[i01] * m01 + p[i11] * m11;
        sd[c] = (_Float16)v;
      }
    }
    __syncthreads();

    // kk = k_w * sampled_k  (WMMA)
    gemm256x16(kw16, s_lds + k * 4096, kk_lds, wave, lane);
    __syncthreads();

    // per-head similarity
    if (t < 64) {
      const int pix = t & 15, h = t >> 4;
      const float* qp = q_lds + pix * 256 + h * 64;
      const float* kp = kk_lds + pix * 256 + h * 64;
      float s = 0.0f;
#pragma unroll 8
      for (int j = 0; j < 64; ++j) s = fmaf(qp[j], kp[j], s);
      sim[(k * NHc + h) * 16 + pix] = s * 0.125f;  // 1/sqrt(64)
    }
    __syncthreads();
  }

  // softmax over K, mean over heads -> w_k
  if (t < 16) {
    const int pix = t;
    float wkk[Kc];
#pragma unroll
    for (int k = 0; k < Kc; ++k) wkk[k] = 0.0f;
#pragma unroll
    for (int h = 0; h < NHc; ++h) {
      float m = -1e30f;
#pragma unroll
      for (int k = 0; k < Kc; ++k) m = fmaxf(m, sim[(k * NHc + h) * 16 + pix]);
      float e[Kc], ssum = 0.0f;
#pragma unroll
      for (int k = 0; k < Kc; ++k) {
        e[k] = __expf(sim[(k * NHc + h) * 16 + pix] - m);
        ssum += e[k];
      }
      float inv = 0.25f / ssum;
#pragma unroll
      for (int k = 0; k < Kc; ++k) wkk[k] += e[k] * inv;
    }
#pragma unroll
    for (int k = 0; k < Kc; ++k) wk[pix * Kc + k] = wkk[k];
  }
  __syncthreads();

  // aligned = sum_k w_k * sampled_k  -> f16 LDS [pix][ch]
  {
    const int pix = t & 15, cbk = (t >> 4) * 16;
    float wl[Kc];
#pragma unroll
    for (int k = 0; k < Kc; ++k) wl[k] = wk[pix * Kc + k];
    _Float16* ad = al16 + pix * 256 + cbk;
#pragma unroll
    for (int c = 0; c < 16; ++c) {
      float a = 0.0f;
#pragma unroll
      for (int k = 0; k < Kc; ++k)
        a = fmaf((float)s_lds[k * 4096 + pix * 256 + cbk + c], wl[k], a);
      ad[c] = (_Float16)a;
    }
  }
  __syncthreads();

  // y = out_w * aligned (WMMA), then diff + per-(b,c) sum/max reductions
  {
    const int n = lane & 15, hi = (lane >> 4) & 1;
    const int kofs = hi ? 8 : 0;
#pragma unroll
    for (int tt = 0; tt < 2; ++tt) {
      const int mt = wave + tt * 8;
      v8f acc = {};
#pragma unroll
      for (int kb = 0; kb < 8; ++kb) {
        const int kbase = kb * 32 + kofs;
        v16h a  = load_frag(ow16 + (size_t)(mt * 16 + n) * 256 + kbase);
        v16h bb = load_frag(al16 + n * 256 + kbase);
        acc = __builtin_amdgcn_wmma_f32_16x16x32_f16(false, a, false, bb,
                                                     (short)0, acc, false, false);
      }
#pragma unroll
      for (int r = 0; r < 8; ++r) {
        const int ch = mt * 16 + r + (hi ? 8 : 0);
        float xv = xmb[(size_t)ch * HWc + pixbase + n];
        float df = xv - acc[r];
        diff_ws[(size_t)b * Cc * HWc + (size_t)ch * HWc + pixbase + n] = df;
        float ssum = df, smax = df;
        for (int m = 1; m < 16; m <<= 1) {
          ssum += __shfl_xor(ssum, m, 32);
          smax = fmaxf(smax, __shfl_xor(smax, m, 32));
        }
        if (n == 0) {
          atomicAdd(&sum_ws[b * Cc + ch], ssum);
          atomicMax(&maxe_ws[b * Cc + ch], enc_f32(smax));
        }
      }
    }
  }
}

// ---------------- kernel 3: gate MLP per batch -----------------------------
__global__ void __launch_bounds__(256)
k3_gate(const float* __restrict__ sum_ws, const unsigned* __restrict__ maxe_ws,
        const float* __restrict__ fc1w, const float* __restrict__ fc1b,
        const float* __restrict__ fc2w, const float* __restrict__ fc2b,
        float* __restrict__ gate) {
  __shared__ float s[2 * Cc];
  __shared__ float hvec[HIDc];
  const int b = blockIdx.x, t = threadIdx.x;
  s[t]      = sum_ws[b * Cc + t] * (1.0f / (float)HWc);
  s[Cc + t] = dec_f32(maxe_ws[b * Cc + t]);
  __syncthreads();
  if (t < HIDc) {
    float a = fc1b[t];
    for (int i = 0; i < 2 * Cc; ++i) a = fmaf(s[i], fc1w[t * 2 * Cc + i], a);
    hvec[t] = fmaxf(a, 0.0f);
  }
  __syncthreads();
  float a = fc2b[t];
#pragma unroll
  for (int j = 0; j < HIDc; ++j) a = fmaf(hvec[j], fc2w[t * HIDc + j], a);
  gate[b * Cc + t] = 1.0f / (1.0f + __expf(-a));
}

// ---------------- kernel 4: residual + gated diff --------------------------
__global__ void __launch_bounds__(256)
k4_final(const float* __restrict__ xm, const float* __restrict__ diff,
         const float* __restrict__ gate, float* __restrict__ out) {
  size_t i = (size_t)blockIdx.x * 256 + threadIdx.x;
  int bc = (int)(i >> 12);  // (b*C + c), HW = 4096
  out[i] = xm[i] + gate[bc] * diff[i];
}

// ---------------- launcher -------------------------------------------------
extern "C" void kernel_launch(void* const* d_in, const int* in_sizes, int n_in,
                              void* d_out, int out_size, void* d_ws, size_t ws_size,
                              hipStream_t stream) {
  const float* x_main   = (const float*)d_in[0];
  const float* x_aux    = (const float*)d_in[1];
  const float* offset_w = (const float*)d_in[2];
  const float* offset_b = (const float*)d_in[3];
  const float* q_w      = (const float*)d_in[4];
  const float* k_w      = (const float*)d_in[5];
  const float* out_w    = (const float*)d_in[6];
  const float* fc1_w    = (const float*)d_in[7];
  const float* fc1_b    = (const float*)d_in[8];
  const float* fc2_w    = (const float*)d_in[9];
  const float* fc2_b    = (const float*)d_in[10];
  float* out = (float*)d_out;

  char* ws = (char*)d_ws;
  _Float16* qw16 = (_Float16*)(ws + 0);                      // 128 KB
  _Float16* kw16 = (_Float16*)(ws + 131072);                 // 128 KB
  _Float16* ow16 = (_Float16*)(ws + 262144);                 // 128 KB
  float*    grid = (float*)(ws + 393216);                    // 2 MB  (B*K*HW*2)
  float*    sumw = (float*)(ws + 2490368);                   // 8 KB
  unsigned* maxe = (unsigned*)(ws + 2498560);                // 8 KB
  float*    gate = (float*)(ws + 2506752);                   // 8 KB
  float*    diff = (float*)(ws + 2514944);                   // 32 MB

  (void)in_sizes; (void)n_in; (void)out_size; (void)ws_size;

  // k0: weight f16 conversion + reduction init
  k0_prep<<<(Cc * Cc + 255) / 256, 256, 0, stream>>>(
      q_w, k_w, out_w, qw16, kw16, ow16, sumw, maxe);

  // k1: offsets conv + grid
  k1_offsets<<<(Bc * HWc) / 256, 256, 0, stream>>>(x_main, offset_w, offset_b, grid);

  // k2: fused deformable attention (WMMA)
  const size_t smem = 8192 + 16384 + 65536 + 16384 + 8192 + 2048 + 512; // 117248 B
  k2_main<<<Bc * (HWc / 16), 256, smem, stream>>>(
      x_main, x_aux, qw16, kw16, ow16, grid, diff, sumw, maxe);

  // k3: gate MLP
  k3_gate<<<Bc, 256, 0, stream>>>(sumw, maxe, fc1_w, fc1_b, fc2_w, fc2_b, gate);

  // k4: residual
  k4_final<<<(Bc * Cc * HWc) / 256, 256, 0, stream>>>(x_main, diff, gate, out);
}
